// LSTM_47253230190667
// MI455X (gfx1250) — matrix operand order repaired and checked by reading, not compiled, not run on hardware
//
#include <hip/hip_runtime.h>

typedef __bf16 bf16_t;
typedef __attribute__((ext_vector_type(16))) __bf16 v16bf;
typedef __attribute__((ext_vector_type(8)))  __bf16 v8bf;
typedef __attribute__((ext_vector_type(8)))  float   v8f;

#define I_DIM 512
#define H_DIM 512
#define B_DIM 64
#define T_DIM 1024
#define G4    2048   // 4*H
#define M1    65536  // B*T
#define NBLK  32     // persistent-kernel grid size (H/16)

// ---------------------------------------------------------------------------
// Convert + transpose one 4-gate weight stack: out is (G4 rows x 512 K cols),
// row n = gate(n/512) output-column (n%512), contiguous K. out[n*512 + k] =
// W_gate[k*512 + (n%512)].  Gate order along N: [f, i, g, o].
// ---------------------------------------------------------------------------
__global__ void prep_weights(const float* __restrict__ Wf, const float* __restrict__ Wi,
                             const float* __restrict__ Wg, const float* __restrict__ Wo,
                             bf16_t* __restrict__ WT) {
    long idx = (long)blockIdx.x * blockDim.x + threadIdx.x;  // over G4*512
    if (idx >= (long)G4 * I_DIM) return;
    int n = (int)(idx / I_DIM);
    int k = (int)(idx % I_DIM);
    int gi = n >> 9;          // 0..3
    int j  = n & 511;
    const float* W = (gi == 0) ? Wf : (gi == 1) ? Wi : (gi == 2) ? Wg : Wo;
    WT[idx] = (bf16_t)W[(long)k * H_DIM + j];
}

__global__ void prep_bias(const float* __restrict__ bf_, const float* __restrict__ bi_,
                          const float* __restrict__ bg_, const float* __restrict__ bo_,
                          const float* __restrict__ hf_, const float* __restrict__ hi_,
                          const float* __restrict__ hg_, const float* __restrict__ ho_,
                          float* __restrict__ bx) {
    int n = blockIdx.x * blockDim.x + threadIdx.x;
    if (n >= G4) return;
    int gi = n >> 9, j = n & 511;
    const float* a = (gi == 0) ? bf_ : (gi == 1) ? bi_ : (gi == 2) ? bg_ : bo_;
    const float* b = (gi == 0) ? hf_ : (gi == 1) ? hi_ : (gi == 2) ? hg_ : ho_;
    bx[n] = a[j] + b[j];
}

__global__ void conv_x_bf16(const float* __restrict__ x, bf16_t* __restrict__ xb, long n) {
    long i = (long)blockIdx.x * blockDim.x + threadIdx.x;
    if (i < n) xb[i] = (bf16_t)x[i];
}

__global__ void init_state(bf16_t* __restrict__ hb0, unsigned* __restrict__ cnt) {
    int i = blockIdx.x * blockDim.x + threadIdx.x;
    if (i == 0) *cnt = 0u;
    if (i < B_DIM * H_DIM) hb0[i] = (bf16_t)0.0f;
}

// ---------------------------------------------------------------------------
// Fragment loaders for 16-bit WMMA operands (wave32), same contiguous k order in A and B.
// A (MxK row-major, contiguous K): lane L holds row M0+L%16,
//   K = Kbase + {0..15}, Kbase = (L<16 ? 0 : 16).
// B (stored N-major = rows are output cols, contiguous K): lane L holds col
//   N0+L%16, K = Kbase + {0..15}, Kbase = (L<16 ? 0 : 16).
// ---------------------------------------------------------------------------
__device__ __forceinline__ v16bf load_fragA(const bf16_t* p /* row base + Kbase */) {
    v8bf lo = *(const v8bf*)(p);
    v8bf hi = *(const v8bf*)(p + 8);
    v16bf a;
#pragma unroll
    for (int e = 0; e < 8; ++e) { a[e] = lo[e]; a[8 + e] = hi[e]; }
    return a;
}

__device__ __forceinline__ v16bf load_fragB(const bf16_t* p) {
    return *(const v16bf*)(p);
}

__device__ __forceinline__ v16bf load_fragB_lds(const bf16_t* p) {
    v8bf lo = *(const v8bf*)(p);
    v8bf hi = *(const v8bf*)(p + 8);
    v16bf b;
#pragma unroll
    for (int e = 0; e < 8; ++e) { b[e] = lo[e]; b[8 + e] = hi[e]; }
    return b;
}

#define WMMA_BF16(ACC, A, Bm) \
    (ACC) = __builtin_amdgcn_wmma_f32_16x16x32_bf16(false, (A), false, (Bm), (short)0, (ACC), false, false)

// ---------------------------------------------------------------------------
// GEMM1: x_proj[m][n] = sum_k xb[m][k] * WxT[n][k] + bx[n]
// Block = 128 threads (4 waves). Wave w -> M-tile (blockIdx.y*4+w), 4 N-tiles
// starting at blockIdx.x*64. K = 512 (16 wmma steps).
// ---------------------------------------------------------------------------
__global__ void gemm_xproj(const bf16_t* __restrict__ A, const bf16_t* __restrict__ BT,
                           const float* __restrict__ bias, float* __restrict__ Cout) {
    const int wave = threadIdx.x >> 5;
    const int lane = threadIdx.x & 31;
    const int lm = lane & 15;
    const int hi = (lane >> 4) & 1;
    const long m0 = ((long)blockIdx.y * 4 + wave) * 16;
    const int  n0 = blockIdx.x * 64;

    const bf16_t* arow = A + (m0 + lm) * I_DIM + (hi ? 16 : 0);
    const bf16_t* brow = BT + (long)(n0 + lm) * I_DIM + (hi ? 16 : 0);

    v8f acc0 = {}, acc1 = {}, acc2 = {}, acc3 = {};
#pragma unroll 4
    for (int k0 = 0; k0 < I_DIM; k0 += 32) {
        __builtin_prefetch(arow + k0 + 128, 0, 1);
        __builtin_prefetch(brow + k0 + 128, 0, 1);
        v16bf a  = load_fragA(arow + k0);
        v16bf b0 = load_fragB(brow + k0);
        v16bf b1 = load_fragB(brow + k0 + 16l * I_DIM);
        v16bf b2 = load_fragB(brow + k0 + 32l * I_DIM);
        v16bf b3 = load_fragB(brow + k0 + 48l * I_DIM);
        WMMA_BF16(acc0, a, b0);
        WMMA_BF16(acc1, a, b1);
        WMMA_BF16(acc2, a, b2);
        WMMA_BF16(acc3, a, b3);
    }

    const float bv0 = bias[n0 + lm];
    const float bv1 = bias[n0 + 16 + lm];
    const float bv2 = bias[n0 + 32 + lm];
    const float bv3 = bias[n0 + 48 + lm];
#pragma unroll
    for (int r = 0; r < 8; ++r) {
        long m = m0 + r + (hi ? 8 : 0);
        float* crow = Cout + m * (long)G4 + n0 + lm;
        crow[0]  = acc0[r] + bv0;
        crow[16] = acc1[r] + bv1;
        crow[32] = acc2[r] + bv2;
        crow[48] = acc3[r] + bv3;
    }
}

// ---------------------------------------------------------------------------
// Persistent LSTM recurrence. Grid = 32 blocks (one hidden tile of 16 cols
// each), block = 128 threads = 4 waves (one 16-row batch tile each).
//  * Wh slice (4 gates x 16 cols x 512 K, bf16 = 64 KB) DMA'd into LDS once
//    via global_load_async_to_lds_b128 (ASYNCcnt path, no VGPR round-trip).
//  * Cell state c lives entirely in registers (each (m,j) owned by one lane).
//  * h round-trips through global bf16 each step (true cross-block dep),
//    double-buffered; cross-block sync via device-scope arrive counter.
// ---------------------------------------------------------------------------
__global__ void lstm_persistent(const bf16_t* __restrict__ WhT,
                                const float* __restrict__ xp,
                                bf16_t* __restrict__ hb0, bf16_t* __restrict__ hb1,
                                float* __restrict__ out,
                                unsigned* __restrict__ barrier_cnt) {
    __shared__ __align__(32) bf16_t sWh[64 * H_DIM];   // 64 KB of 320 KB WGP LDS

    const int wave = threadIdx.x >> 5;
    const int lane = threadIdx.x & 31;
    const int lm = lane & 15;
    const int hi = (lane >> 4) & 1;
    const int j0 = blockIdx.x * 16;     // this block's hidden tile
    const int m0 = wave * 16;           // this wave's batch tile
    const int j  = j0 + lm;

    // --- async-DMA the Wh slice into LDS: sWh[(g*16 + jc)*512 + k].
    // Per-lane global_load_async_to_lds_b128: 16 B each, 4096 transfers total.
    // LDS byte address = low 32 bits of the generic pointer (LDS aperture).
    for (int idx = threadIdx.x; idx < 64 * H_DIM / 8; idx += blockDim.x) {
        int row = idx >> 6;           // 0..63  (g*16 + jc)
        int chunk = idx & 63;         // 8-element (16 B) chunks of the row
        int g = row >> 4, jc = row & 15;
        const bf16_t* src = WhT + (long)(g * 512 + j0 + jc) * H_DIM + chunk * 8;
        unsigned ldsoff = (unsigned)(unsigned long long)(sWh + row * H_DIM + chunk * 8);
        asm volatile("global_load_async_to_lds_b128 %0, %1, off"
                     :: "v"(ldsoff), "v"(src) : "memory");
    }
    asm volatile("s_wait_asynccnt 0" ::: "memory");
    __syncthreads();

    const bf16_t* sF = sWh + (0 * 16 + lm) * H_DIM + (hi ? 16 : 0);
    const bf16_t* sI = sWh + (1 * 16 + lm) * H_DIM + (hi ? 16 : 0);
    const bf16_t* sG = sWh + (2 * 16 + lm) * H_DIM + (hi ? 16 : 0);
    const bf16_t* sO = sWh + (3 * 16 + lm) * H_DIM + (hi ? 16 : 0);

    float creg[8];
#pragma unroll
    for (int r = 0; r < 8; ++r) creg[r] = 0.0f;

    for (int t = 0; t < T_DIM; ++t) {
        const bf16_t* hin  = (t & 1) ? hb1 : hb0;
        bf16_t*       hout = (t & 1) ? hb0 : hb1;
        const bf16_t* arow = hin + (long)(m0 + lm) * H_DIM + (hi ? 16 : 0);

        v8f accF = {}, accI = {}, accG = {}, accO = {};
#pragma unroll 4
        for (int k0 = 0; k0 < H_DIM; k0 += 32) {
            v16bf a  = load_fragA(arow + k0);
            v16bf vf = load_fragB_lds(sF + k0);
            v16bf vi = load_fragB_lds(sI + k0);
            v16bf vg = load_fragB_lds(sG + k0);
            v16bf vo = load_fragB_lds(sO + k0);
            WMMA_BF16(accF, a, vf);
            WMMA_BF16(accI, a, vi);
            WMMA_BF16(accG, a, vg);
            WMMA_BF16(accO, a, vo);
        }

#pragma unroll
        for (int r = 0; r < 8; ++r) {
            int  m   = m0 + r + (hi ? 8 : 0);          // batch index 0..63
            long row = ((long)m * T_DIM + t) * G4;     // x is (B,T,*) flattened
            float gf = accF[r] + xp[row + 0 * 512 + j];
            float gi = accI[r] + xp[row + 1 * 512 + j];
            float gg = accG[r] + xp[row + 2 * 512 + j];
            float go = accO[r] + xp[row + 3 * 512 + j];
            float f = 1.0f / (1.0f + __expf(-gf));
            float i = 1.0f / (1.0f + __expf(-gi));
            float g = tanhf(gg);
            float o = 1.0f / (1.0f + __expf(-go));
            float cn = f * creg[r] + i * g;
            creg[r] = cn;
            float h = o * tanhf(cn);
            long cidx = (long)m * H_DIM + j;
            hout[cidx] = (bf16_t)h;
            if (t == T_DIM - 1) {                      // final outputs: (h, c)
                out[cidx] = h;
                out[B_DIM * H_DIM + cidx] = cn;
            }
        }

        // prefetch next step's x_proj slice while waiting at the barrier
        if (t + 1 < T_DIM) {
#pragma unroll
            for (int r = 0; r < 8; ++r) {
                long rown = ((long)(m0 + r + (hi ? 8 : 0)) * T_DIM + (t + 1)) * G4;
                __builtin_prefetch(xp + rown + j, 0, 0);
                __builtin_prefetch(xp + rown + 1024 + j, 0, 0);
            }
        }

        // device-scope grid barrier (monotonic counter, target = NBLK*(t+1))
        __threadfence();
        __syncthreads();
        if (threadIdx.x == 0) {
            __hip_atomic_fetch_add(barrier_cnt, 1u, __ATOMIC_ACQ_REL,
                                   __HIP_MEMORY_SCOPE_AGENT);
            const unsigned target = (unsigned)NBLK * (unsigned)(t + 1);
            while (__hip_atomic_load(barrier_cnt, __ATOMIC_ACQUIRE,
                                     __HIP_MEMORY_SCOPE_AGENT) < target) {
                __builtin_amdgcn_s_sleep(1);
            }
        }
        __syncthreads();
        __threadfence();
    }
}

// ---------------------------------------------------------------------------
extern "C" void kernel_launch(void* const* d_in, const int* in_sizes, int n_in,
                              void* d_out, int out_size, void* d_ws, size_t ws_size,
                              hipStream_t stream) {
    const float* x    = (const float*)d_in[0];
    const float* W_ii = (const float*)d_in[1];
    const float* W_hi = (const float*)d_in[2];
    const float* W_if = (const float*)d_in[3];
    const float* W_hf = (const float*)d_in[4];
    const float* W_ig = (const float*)d_in[5];
    const float* W_hg = (const float*)d_in[6];
    const float* W_io = (const float*)d_in[7];
    const float* W_ho = (const float*)d_in[8];
    const float* b_ii = (const float*)d_in[9];
    const float* b_hi = (const float*)d_in[10];
    const float* b_if = (const float*)d_in[11];
    const float* b_hf = (const float*)d_in[12];
    const float* b_ig = (const float*)d_in[13];
    const float* b_hg = (const float*)d_in[14];
    const float* b_io = (const float*)d_in[15];
    const float* b_ho = (const float*)d_in[16];

    char* ws = (char*)d_ws;
    // Workspace layout (all offsets 4 KiB aligned):
    bf16_t*   xb  = (bf16_t*)(ws);                      //  64 MiB  x in bf16
    bf16_t*   WxT = (bf16_t*)(ws + 67108864);           //   2 MiB  (2048 x 512) N-major
    bf16_t*   WhT = (bf16_t*)(ws + 69206016);           //   2 MiB
    float*    bx  = (float*) (ws + 71303168);           //   8 KiB  fused bias
    bf16_t*   hb0 = (bf16_t*)(ws + 71311360);           //  64 KiB  h bf16 (ping)
    bf16_t*   hb1 = (bf16_t*)(ws + 71376896);           //  64 KiB  h bf16 (pong)
    unsigned* cnt = (unsigned*)(ws + 71442432);         //   4 B    barrier counter
    float*    xp  = (float*) (ws + 71704576);           // 512 MiB  x_proj fp32

    // --- prep: weights (transposed, bf16), fused bias, x in bf16, state zero
    {
        long nW = (long)G4 * I_DIM;
        int blk = 256, grd = (int)((nW + blk - 1) / blk);
        prep_weights<<<grd, blk, 0, stream>>>(W_if, W_ii, W_ig, W_io, WxT);
        prep_weights<<<grd, blk, 0, stream>>>(W_hf, W_hi, W_hg, W_ho, WhT);
    }
    prep_bias<<<(G4 + 255) / 256, 256, 0, stream>>>(b_if, b_ii, b_ig, b_io,
                                                    b_hf, b_hi, b_hg, b_ho, bx);
    {
        long nx = (long)M1 * I_DIM;
        conv_x_bf16<<<(int)((nx + 255) / 256), 256, 0, stream>>>(x, xb, nx);
    }
    init_state<<<(B_DIM * H_DIM + 255) / 256, 256, 0, stream>>>(hb0, cnt);

    // --- phase 1: x_proj = x @ Wx + bx   (bf16 WMMA, f32 accumulate)
    {
        dim3 grid(G4 / 64, (M1 / 16) / 4);   // 32 x 1024 blocks
        gemm_xproj<<<grid, 128, 0, stream>>>(xb, WxT, bx, xp);
    }

    // --- phase 2: persistent recurrence, all 1024 steps in one kernel
    lstm_persistent<<<NBLK, 128, 0, stream>>>(WhT, xp, hb0, hb1,
                                              (float*)d_out, cnt);
}
